// gwnet_54683523612764
// MI455X (gfx1250) — matrix operand order repaired and checked by reading, not compiled, not run on hardware
//
#include <hip/hip_runtime.h>
#include <hip/hip_bf16.h>
#include <math.h>

typedef __bf16 bf16_t;
typedef __attribute__((ext_vector_type(16))) __bf16 v16bf;
typedef __attribute__((ext_vector_type(8)))  float  v8f;

#define NB 64
#define NV 207
#define TT 12
#define RC 32
#define SCC 256
#define ECC 512
#define HC 224
#define NL 8
#define PADN 224
#define POS (NB*NV)          // 13248, multiple of 16

static __device__ __forceinline__ float sigm(float x) { return 1.0f / (1.0f + __expf(-x)); }

static __device__ __forceinline__ v8f wmma_bf(v16bf a, v16bf b, v8f c) {
  return __builtin_amdgcn_wmma_f32_16x16x32_bf16(false, a, false, b, (short)0, c, false, false);
}

// A-tile (16xK slice, row-major, contiguous K). Lane l: row = l&15, khalf = l>>4,
// elements j<8 -> K = kb + khalf*8 + j ; j>=8 -> K = kb + 16 + khalf*8 + (j-8)
static __device__ __forceinline__ v16bf load_a_cont(const bf16_t* A, int lda, int row_base,
                                                    int Mact, int kb, int lane) {
  int m = row_base + (lane & 15);
  if (m >= Mact) m = Mact - 1;
  const bf16_t* r = A + (size_t)m * lda;
  int khalf = lane >> 4;
  int k0 = kb + khalf * 8;
  int k1 = kb + 16 + khalf * 8;
  v16bf v;
#pragma unroll
  for (int j = 0; j < 8; ++j) { v[j] = r[k0 + j]; v[8 + j] = r[k1 + j]; }
  return v;
}

// B-tile from Bt[N][ldb] (i.e. B^T, contiguous K per output column).
// Lane l: col = l&15, elements j<8 hold K = kb + (l>>4)*8 + j ; j>=8 -> K = kb + 16 + (l>>4)*8 + (j-8)
static __device__ __forceinline__ v16bf load_b_cont(const bf16_t* Bt, int ldb, int col_base,
                                                    int Nact, int kb, int lane) {
  int n = col_base + (lane & 15);
  if (n >= Nact) n = Nact - 1;
  const bf16_t* r = Bt + (size_t)n * ldb + kb + (lane >> 4) * 8;
  v16bf v;
#pragma unroll
  for (int j = 0; j < 8; ++j) { v[j] = r[j]; v[8 + j] = r[16 + j]; }
  return v;
}

// ---------------------------------------------------------------------------
// Generic WMMA GEMM: D = A(MxK) * Bt(NxK)^T, one wave per 16x16 tile.
// MODE 0: outf = d(+bias)        MODE 1: outf += d+bias (skip accumulation)
// MODE 2: outb = bf16(relu(d+bias))
// MODE 3: end2 -> dout[(b*12+n)*207+v]
// MODE 4: gcn  -> +residual, write tmp f32, lane-reduced atomic BN stats
// ---------------------------------------------------------------------------
template <int MODE>
__global__ void k_gemm(const bf16_t* __restrict__ A, int lda, long aS,
                       const bf16_t* __restrict__ Bt, int ldb, long bS,
                       int M, int N, int K, const float* __restrict__ bias,
                       float* outf, int ldo, long oS, bf16_t* outb,
                       const float* resid, float* tmpo, float* stats,
                       int T_in, int T_out, int dil, float* dout) {
  int lane = threadIdx.x;
  int mt = blockIdx.x, nt = blockIdx.y, z = blockIdx.z;
  A += (size_t)z * aS;
  Bt += (size_t)z * bS;
  v8f d = {0.f, 0.f, 0.f, 0.f, 0.f, 0.f, 0.f, 0.f};
  for (int kb = 0; kb < K; kb += 32) {
    v16bf a = load_a_cont(A, lda, mt << 4, M, kb, lane);
    v16bf b = load_b_cont(Bt, ldb, nt << 4, N, kb, lane);
    d = wmma_bf(a, b, d);
  }
  int nl = lane & 15;
  int mb = (lane >> 4) * 8;
  int n = (nt << 4) + nl;
  float ssum = 0.f, ssq = 0.f;
#pragma unroll
  for (int r = 0; r < 8; ++r) {
    int m = (mt << 4) + mb + r;
    if (m >= M || n >= N) continue;
    float val = d[r] + (bias ? bias[n] : 0.f);
    if (MODE == 0) {
      outf[(size_t)z * oS + (size_t)m * ldo + n] = val;
    } else if (MODE == 1) {
      outf[(size_t)m * ldo + n] += val;
    } else if (MODE == 2) {
      outb[(size_t)m * ldo + n] = (bf16_t)fmaxf(val, 0.f);
    } else if (MODE == 3) {
      int b2 = m / NV, v2 = m % NV;
      dout[((size_t)b2 * 12 + n) * NV + v2] = val;
    } else {
      int t = m % T_out;
      int rr = m / T_out;
      int v2 = rr % NV, b2 = rr / NV;
      float rv = resid[((size_t)(b2 * NV + v2) * T_in + t + dil) * RC + n];
      float o = val + rv;
      tmpo[(size_t)m * RC + n] = o;
      ssum += o;
      ssq += o * o;
    }
  }
  if (MODE == 4 && n < N) {
    atomicAdd(&stats[n], ssum);
    atomicAdd(&stats[32 + n], ssq);
  }
}

// ---------------------------------------------------------------------------
// TCN: dilated (1,2) filter+gate conv, K=64, packed W [64][64]; fused tanh*sigm.
// Writes h channels 0..31 (channel-contiguous, for skip/gcn GEMMs) AND the
// node-contiguous shadow hv0[b][t][c][v] (for diffusion-hop A operands).
// p = ((b*NV+v)*T_out + t)
// ---------------------------------------------------------------------------
__global__ void k_tcn(const bf16_t* __restrict__ xbf, int T_in, int T_out, int dil,
                      const bf16_t* __restrict__ Wfg, const float* __restrict__ fb,
                      const float* __restrict__ gb, bf16_t* __restrict__ h,
                      bf16_t* __restrict__ hv0) {
  int lane = threadIdx.x;
  int pt = blockIdx.x;
  int m = (pt << 4) | (lane & 15);
  int khalf = lane >> 4;
  int t = m % T_out;
  int r1 = m / T_out;
  int v = r1 % NV, b = r1 / NV;
  const bf16_t* xrow = xbf + (size_t)(b * NV + v) * T_in * RC;
  v8f d0 = {0.f,0.f,0.f,0.f,0.f,0.f,0.f,0.f}, d1 = d0, d2 = d0, d3 = d0;
  for (int kb = 0; kb < 64; kb += 32) {
    int s0 = kb + khalf * 8;
    int s1 = kb + 16 + khalf * 8;
    const bf16_t* p0 = xrow + (size_t)(t + (s0 >= 32 ? dil : 0)) * RC + (s0 & 31);
    const bf16_t* p1 = xrow + (size_t)(t + (s1 >= 32 ? dil : 0)) * RC + (s1 & 31);
    v16bf a;
#pragma unroll
    for (int j = 0; j < 8; ++j) { a[j] = p0[j]; a[8 + j] = p1[j]; }
    v16bf b0 = load_b_cont(Wfg, 64, 0, 64, kb, lane);
    v16bf b1 = load_b_cont(Wfg, 64, 16, 64, kb, lane);
    v16bf b2 = load_b_cont(Wfg, 64, 32, 64, kb, lane);
    v16bf b3 = load_b_cont(Wfg, 64, 48, 64, kb, lane);
    d0 = wmma_bf(a, b0, d0);
    d1 = wmma_bf(a, b1, d1);
    d2 = wmma_bf(a, b2, d2);
    d3 = wmma_bf(a, b3, d3);
  }
  int nl = lane & 15;
  int mb = 8 * khalf;
#pragma unroll
  for (int r = 0; r < 8; ++r) {
    int mm = (pt << 4) + mb + r;
    int tt2 = mm % T_out;
    int rr = mm / T_out;
    int vv = rr % NV, bb = rr / NV;
    size_t base = ((size_t)(bb * NV + vv) * T_out + tt2) * HC;
    size_t vbase = ((size_t)(bb * T_out + tt2) * RC) * PADN + vv;
    float f0 = d0[r] + fb[nl];
    float g0 = d2[r] + gb[nl];
    float o0 = tanhf(f0) * sigm(g0);
    h[base + nl] = (bf16_t)o0;
    hv0[vbase + (size_t)nl * PADN] = (bf16_t)o0;
    float f1 = d1[r] + fb[16 + nl];
    float g1 = d3[r] + gb[16 + nl];
    float o1 = tanhf(f1) * sigm(g1);
    h[base + 16 + nl] = (bf16_t)o1;
    hv0[vbase + (size_t)(16 + nl) * PADN] = (bf16_t)o1;
  }
}

// ---------------------------------------------------------------------------
// GCN diffusion hop: y[b,w,t,c] = sum_v hv_in[b,t,c,v] * A[v,w], per batch b.
// A operand is node-contiguous (hv layout [b][t][c][v_pad224]) -> b128 loads.
// M = (t,c) = 32*T rows, K = v (padded 224), N = w.
// Writes h[ch_out..] (channel-contiguous) and optionally hv_out (next hop's A).
// ---------------------------------------------------------------------------
__global__ void k_hop(bf16_t* __restrict__ h, const bf16_t* __restrict__ hv_in,
                      bf16_t* __restrict__ hv_out, int T, int ch_out,
                      const bf16_t* __restrict__ At) {
  int lane = threadIdx.x;
  int mt = blockIdx.x, nt = blockIdx.y, b = blockIdx.z;
  const bf16_t* Ab = hv_in + (size_t)b * T * RC * PADN;
  int M = T * RC;
  v8f d = {0.f,0.f,0.f,0.f,0.f,0.f,0.f,0.f};
  for (int kb = 0; kb < PADN; kb += 32) {
    v16bf a = load_a_cont(Ab, PADN, mt << 4, M, kb, lane);
    v16bf bb = load_b_cont(At, PADN, nt << 4, PADN, kb, lane);
    d = wmma_bf(a, bb, d);
  }
  int nl = lane & 15;
  int mb = 8 * (lane >> 4);
  int w = (nt << 4) + nl;
  if (w >= NV) return;
#pragma unroll
  for (int r = 0; r < 8; ++r) {
    int mm = (mt << 4) + mb + r;
    int tt = mm >> 5;
    int cc = mm & 31;
    bf16_t o = (bf16_t)d[r];
    h[((size_t)(b * NV + w) * T + tt) * HC + ch_out + cc] = o;
    if (hv_out) hv_out[((size_t)(b * T + tt) * RC + cc) * PADN + w] = o;
  }
}

// ----------------------------- small kernels --------------------------------
__global__ void k_pack_sup(bf16_t* dst, const float* src) {  // dst[w][v]=src[v][w], 224x224
  int idx = blockIdx.x * blockDim.x + threadIdx.x;
  if (idx >= PADN * PADN) return;
  int w = idx / PADN, v = idx % PADN;
  dst[idx] = (v < NV && w < NV) ? (bf16_t)src[v * NV + w] : (bf16_t)0.f;
}

__global__ void k_adp(const float* nv1, const float* nv2, bf16_t* At) {
  int v = blockIdx.x * blockDim.x + threadIdx.x;
  if (v >= NV) return;
  float row[NV];
  float mx = -1e30f;
  for (int w = 0; w < NV; ++w) {
    float s = 0.f;
    for (int d = 0; d < 10; ++d) s += nv1[v * 10 + d] * nv2[d * NV + w];
    s = fmaxf(s, 0.f);
    row[w] = s;
    mx = fmaxf(mx, s);
  }
  float den = 0.f;
  for (int w = 0; w < NV; ++w) { row[w] = __expf(row[w] - mx); den += row[w]; }
  for (int w = 0; w < NV; ++w) At[w * PADN + v] = (bf16_t)(row[w] / den);
}

__global__ void k_pack_wfg(bf16_t* dst, const float* fw, const float* gw) {
  int idx = blockIdx.x * blockDim.x + threadIdx.x;
  if (idx >= NL * 64 * 64) return;
  int k = idx & 63;
  int r = (idx >> 6) & 63;
  int i = idx >> 12;
  float val;
  if (r < 32) val = fw[((i * 32 + r) * 32 + (k & 31)) * 2 + (k >> 5)];
  else        val = gw[((i * 32 + (r - 32)) * 32 + (k & 31)) * 2 + (k >> 5)];
  dst[idx] = (bf16_t)val;
}

__global__ void k_pack_mat(bf16_t* dst, const float* src, int rd, int rs, int cd, int cs) {
  int idx = blockIdx.x * blockDim.x + threadIdx.x;
  if (idx >= rd * cd) return;
  int r = idx / cd, c = idx % cd;
  dst[idx] = (r < rs && c < cs) ? (bf16_t)src[r * cs + c] : (bf16_t)0.f;
}

__global__ void k_start(const float* in, const float* sw, const float* sb, const float* cent,
                        float* xf, bf16_t* xbf) {
  int idx = blockIdx.x * blockDim.x + threadIdx.x;
  if (idx >= NB * NV * 13 * RC) return;
  int c = idx % RC;
  int r = idx / RC;
  int t = r % 13; r /= 13;
  int v = r % NV;
  int b = r / NV;
  float val = sb[c] + cent[v];
  if (t > 0) {
    for (int ci = 0; ci < 2; ++ci)
      val += sw[c * 2 + ci] * in[((size_t)(b * 2 + ci) * NV + v) * TT + (t - 1)];
  }
  xf[idx] = val;
  xbf[idx] = (bf16_t)val;
}

__global__ void k_bn(const float* tmp, const float* stats, const float* g, const float* bb,
                     float cnt, int total, float* xf, bf16_t* xbf) {
  int idx = blockIdx.x * blockDim.x + threadIdx.x;
  if (idx >= total) return;
  int c = idx & 31;
  float m = stats[c] / cnt;
  float var = stats[32 + c] / cnt - m * m;
  float o = (tmp[idx] - m) * rsqrtf(var + 1e-5f) * g[c] + bb[c];
  xf[idx] = o;
  xbf[idx] = (bf16_t)o;
}

__global__ void k_relu_ln(const float* skip, const float* lg, const float* lb, float* xln) {
  int b = blockIdx.x;
  int tid = threadIdx.x;
  __shared__ float red[256], red2[256];
  const float* sb = skip + (size_t)b * (NV * SCC);
  float s = 0.f, ss = 0.f;
  for (int e = tid; e < NV * SCC; e += 256) {
    float v = fmaxf(sb[e], 0.f);
    s += v; ss += v * v;
  }
  red[tid] = s; red2[tid] = ss;
  __syncthreads();
  for (int st = 128; st > 0; st >>= 1) {
    if (tid < st) { red[tid] += red[tid + st]; red2[tid] += red2[tid + st]; }
    __syncthreads();
  }
  float m = red[0] / (float)(NV * SCC);
  float var = red2[0] / (float)(NV * SCC) - m * m;
  float inv = rsqrtf(var + 1e-5f);
  for (int e = tid; e < NV * SCC; e += 256) {
    int v = e >> 8, c = e & 255;
    float x = fmaxf(sb[e], 0.f);
    xln[(size_t)b * (NV * SCC) + e] = (x - m) * inv * lg[c * NV + v] + lb[c * NV + v];
  }
}

// --- temporal attention (faithful sloppy reshapes) ---
__global__ void k_ta1(const float* in, const float* u1, float* vec) {
  int q = blockIdx.x * blockDim.x + threadIdx.x;
  if (q >= 1536) return;
  float acc = 0.f;
  for (int n = 0; n < NV; ++n) {
    int F = n * 1536 + q;
    int ci = F & 1; F >>= 1;
    int v = F % NV; F /= NV;
    int t = F % TT; int b = F / TT;
    acc += in[((size_t)(b * 2 + ci) * NV + v) * TT + t] * u1[n];
  }
  vec[q] = acc;
}

__global__ void k_ta1b(const float* vec, const float* u2, float* lhs) {
  int idx = blockIdx.x * blockDim.x + threadIdx.x;
  if (idx >= NB * TT * NV) return;
  int n = idx % NV;
  int r = idx / NV;  // b*12+t
  lhs[idx] = vec[r * 2] * u2[n] + vec[r * 2 + 1] * u2[NV + n];
}

__global__ void k_ta2(const float* in, const float* u3, float* rhsbuf) {
  int e = blockIdx.x * blockDim.x + threadIdx.x;
  if (e >= NV * NB * TT) return;
  float acc = 0.f;
  int t0 = e % TT;
  int r = e / TT;
  int b0 = r % NB;
  int n0 = r / NB;
  for (int ci = 0; ci < 2; ++ci) {
    int g = ((n0 * NB + b0) * TT + t0) * 2 + ci;
    int tt = g % TT; g /= TT;
    int cc = g % 2;  g /= 2;
    int bb = g % NB; int nn = g / NB;
    acc += in[((size_t)(bb * 2 + cc) * NV + nn) * TT + tt] * u3[ci];
  }
  rhsbuf[e] = acc;
}

__global__ void k_ta3(const float* lhs, const float* rhs, const float* be, const float* ve,
                      const float* hx, float* ttt) {
  int b = blockIdx.x;
  int tid = threadIdx.x;
  __shared__ float sSig[144], sE[144], sS[144];
  if (tid < 144) {
    int i = tid / 12, j = tid % 12;
    float a = 0.f;
    for (int n = 0; n < NV; ++n)
      a += lhs[((size_t)b * 12 + i) * NV + n] * rhs[((size_t)b * NV + n) * 12 + j];
    sSig[tid] = sigm(a + be[tid]);
  }
  __syncthreads();
  if (tid < 144) {
    int i = tid / 12, j = tid % 12;
    float a = 0.f;
    for (int k = 0; k < 12; ++k) a += ve[i * 12 + k] * sSig[k * 12 + j];
    sE[tid] = a;
  }
  __syncthreads();
  if (tid < 12) {
    int j = tid;
    float m = -1e30f;
    for (int i = 0; i < 12; ++i) m = fmaxf(m, sE[i * 12 + j]);
    float s = 0.f;
    for (int i = 0; i < 12; ++i) { float e = __expf(sE[i * 12 + j] - m); sS[i * 12 + j] = e; s += e; }
    for (int i = 0; i < 12; ++i) sS[i * 12 + j] /= s;
  }
  __syncthreads();
  for (int e = tid; e < 12 * NV; e += blockDim.x) {
    int i = e / NV, v = e % NV;
    float a = 0.f;
    for (int j = 0; j < 12; ++j) a += sS[i * 12 + j] * hx[(size_t)b * (12 * NV) + j * NV + v];
    ttt[(size_t)b * (12 * NV) + e] = a;
  }
}

// --- spatial attention ---
__global__ void k_sa_lhs(const float* in, const float* w1, const float* w2, float* lhs) {
  int idx = blockIdx.x * blockDim.x + threadIdx.x;
  if (idx >= NB * NV * TT) return;
  int tp = idx % TT;
  int r = idx / TT;
  int n = r % NV, b = r / NV;
  float acc = 0.f;
  for (int ci = 0; ci < 2; ++ci) {
    float pre = 0.f;
    for (int t = 0; t < TT; ++t) pre += in[((size_t)(b * NV + n) * 2 + ci) * TT + t] * w1[t];
    acc += pre * w2[ci * TT + tp];
  }
  lhs[idx] = acc;
}

__global__ void k_sa_rhs(const float* in, const float* w3, float* rhs) {
  int idx = blockIdx.x * blockDim.x + threadIdx.x;
  if (idx >= NB * TT * NV) return;
  int n = idx % NV;
  int r = idx / NV;
  int t = r % TT, b = r / TT;
  float acc = 0.f;
  for (int ci = 0; ci < 2; ++ci)
    acc += in[((size_t)(ci * NB + b) * TT + t) * NV + n] * w3[ci];
  rhs[idx] = acc;
}

__global__ void k_sa2(const float* lhs, const float* rhs, const float* bs, bf16_t* Pt) {
  int idx = blockIdx.x * blockDim.x + threadIdx.x;
  if (idx >= NB * NV * PADN) return;
  int k = idx % PADN;
  int r = idx / PADN;
  int j = r % NV, b = r / NV;
  float val = 0.f;
  if (k < NV) {
    float acc = 0.f;
    for (int t = 0; t < TT; ++t)
      acc += lhs[((size_t)b * NV + k) * TT + t] * rhs[((size_t)b * TT + t) * NV + j];
    val = sigm(acc + bs[k * NV + j]);
  }
  Pt[idx] = (bf16_t)val;
}

__global__ void k_soft_s(float* S) {  // softmax over axis=1 (i) of (B,207,207)
  int idx = blockIdx.x * blockDim.x + threadIdx.x;
  if (idx >= NB * NV) return;
  int j = idx % NV, b = idx / NV;
  float* base = S + (size_t)b * NV * NV + j;
  float m = -1e30f;
  for (int i = 0; i < NV; ++i) m = fmaxf(m, base[i * NV]);
  float s = 0.f;
  for (int i = 0; i < NV; ++i) s += __expf(base[i * NV] - m);
  float inv = 1.f / s;
  for (int i = 0; i < NV; ++i) base[i * NV] = __expf(base[i * NV] - m) * inv;
}

__global__ void k_concat(const float* Sn, const float* ttt, bf16_t* cat) {
  int idx = blockIdx.x * blockDim.x + threadIdx.x;
  if (idx >= POS * PADN) return;
  int jj = idx % PADN;
  int r = idx / PADN;
  int n = r % NV, b = r / NV;
  float val = 0.f;
  if (jj < 220) {
    int q = n * 220 + jj;
    int rr = q / NV, cc = q % NV;
    if (rr < NV)        val = Sn[(size_t)b * NV * NV + rr * NV + cc];
    else if (rr < 219)  val = ttt[(size_t)b * (12 * NV) + (rr - 207) * NV + cc];
  }
  cat[idx] = (bf16_t)val;
}

__global__ void k_addlin(const float* xln, const float* lin, const float* twe, bf16_t* xend) {
  int idx = blockIdx.x * blockDim.x + threadIdx.x;
  if (idx >= POS * SCC) return;
  int c = idx & 255;
  int p = idx >> 8;
  int b = p / NV, v = p % NV;
  float val = xln[idx] + twe[0] * lin[(size_t)b * (NV * SCC) + c * NV + v];
  xend[idx] = (bf16_t)val;
}

// ---------------------------------------------------------------------------
#define CDIV(a, b) (((a) + (b)-1) / (b))

extern "C" void kernel_launch(void* const* d_in, const int* in_sizes, int n_in,
                              void* d_out, int out_size, void* d_ws, size_t ws_size,
                              hipStream_t stream) {
  (void)in_sizes; (void)n_in; (void)out_size; (void)ws_size;
  const float* in       = (const float*)d_in[0];
  const float* support0 = (const float*)d_in[1];
  const float* support1 = (const float*)d_in[2];
  const float* start_w  = (const float*)d_in[3];
  const float* start_b  = (const float*)d_in[4];
  const float* cent     = (const float*)d_in[5];
  const float* nodevec1 = (const float*)d_in[6];
  const float* nodevec2 = (const float*)d_in[7];
  const float* filter_w = (const float*)d_in[8];
  const float* filter_b = (const float*)d_in[9];
  const float* gate_w   = (const float*)d_in[10];
  const float* gate_b   = (const float*)d_in[11];
  const float* skip_w   = (const float*)d_in[12];
  const float* skip_b   = (const float*)d_in[13];
  const float* gcn_w    = (const float*)d_in[14];
  const float* gcn_b    = (const float*)d_in[15];
  const float* bn_g     = (const float*)d_in[16];
  const float* bn_b     = (const float*)d_in[17];
  const float* ln_g     = (const float*)d_in[18];
  const float* ln_b     = (const float*)d_in[19];
  const float* sa_w1    = (const float*)d_in[20];
  const float* sa_w2    = (const float*)d_in[21];
  const float* sa_w3    = (const float*)d_in[22];
  const float* sa_bs    = (const float*)d_in[23];
  const float* sa_vs    = (const float*)d_in[24];
  const float* ta_u1    = (const float*)d_in[25];
  const float* ta_u2    = (const float*)d_in[26];
  const float* ta_u3    = (const float*)d_in[27];
  const float* ta_be    = (const float*)d_in[28];
  const float* ta_ve    = (const float*)d_in[29];
  const float* h_x_sp   = (const float*)d_in[30];
  const float* t_we     = (const float*)d_in[31];
  const float* lin_w    = (const float*)d_in[32];
  const float* lin_b    = (const float*)d_in[33];
  const float* end1_w   = (const float*)d_in[34];
  const float* end1_b   = (const float*)d_in[35];
  const float* end2_w   = (const float*)d_in[36];
  const float* end2_b   = (const float*)d_in[37];
  float* dout = (float*)d_out;

  char* ws = (char*)d_ws;
  size_t off = 0;
  auto alloc = [&](size_t bytes) -> void* {
    off = (off + 255) & ~(size_t)255;
    void* p = ws + off;
    off += bytes;
    return p;
  };
  const size_t hvRegion = (size_t)NB * TT * RC * PADN;  // per hv region (elements)
  float*  xf     = (float*)alloc((size_t)NB * NV * 13 * RC * 4);
  bf16_t* xbf    = (bf16_t*)alloc((size_t)NB * NV * 13 * RC * 2);
  bf16_t* h      = (bf16_t*)alloc((size_t)NB * NV * 12 * HC * 2);
  bf16_t* hv     = (bf16_t*)alloc(hvRegion * 2 * 2);          // two regions
  float*  tmp    = (float*)alloc((size_t)NB * NV * 12 * RC * 4);
  float*  stats  = (float*)alloc(64 * 4);
  float*  skip   = (float*)alloc((size_t)POS * SCC * 4);
  float*  xln    = (float*)alloc((size_t)POS * SCC * 4);
  float*  linbuf = (float*)alloc((size_t)POS * SCC * 4);
  bf16_t* catb   = (bf16_t*)alloc((size_t)POS * PADN * 2);
  bf16_t* xend   = (bf16_t*)alloc((size_t)POS * SCC * 2);
  bf16_t* e1b    = (bf16_t*)alloc((size_t)POS * ECC * 2);
  bf16_t* At0    = (bf16_t*)alloc(PADN * PADN * 2);
  bf16_t* At1    = (bf16_t*)alloc(PADN * PADN * 2);
  bf16_t* Atad   = (bf16_t*)alloc(PADN * PADN * 2);
  bf16_t* Wfg    = (bf16_t*)alloc((size_t)NL * 64 * 64 * 2);
  bf16_t* Wsk    = (bf16_t*)alloc((size_t)NL * SCC * RC * 2);
  bf16_t* Wgc    = (bf16_t*)alloc((size_t)NL * RC * HC * 2);
  bf16_t* We1    = (bf16_t*)alloc((size_t)ECC * SCC * 2);
  bf16_t* We2    = (bf16_t*)alloc((size_t)16 * ECC * 2);
  bf16_t* Wli    = (bf16_t*)alloc((size_t)SCC * PADN * 2);
  bf16_t* vsb    = (bf16_t*)alloc((size_t)NV * PADN * 2);
  bf16_t* Ptb    = (bf16_t*)alloc((size_t)NB * NV * PADN * 2);
  float*  Smat   = (float*)alloc((size_t)NB * NV * NV * 4);
  float*  vec    = (float*)alloc(1536 * 4);
  float*  lhs_t  = (float*)alloc((size_t)NB * TT * NV * 4);
  float*  rhs_t  = (float*)alloc((size_t)NB * TT * NV * 4);
  float*  tttb   = (float*)alloc((size_t)NB * TT * NV * 4);
  float*  lhs_s  = (float*)alloc((size_t)NB * NV * TT * 4);
  float*  rhs_s  = (float*)alloc((size_t)NB * TT * NV * 4);

  // ---- one-time packing ----
  k_pack_sup<<<CDIV(PADN * PADN, 256), 256, 0, stream>>>(At0, support0);
  k_pack_sup<<<CDIV(PADN * PADN, 256), 256, 0, stream>>>(At1, support1);
  hipMemsetAsync(Atad, 0, PADN * PADN * 2, stream);
  k_adp<<<CDIV(NV, 64), 64, 0, stream>>>(nodevec1, nodevec2, Atad);
  k_pack_wfg<<<CDIV(NL * 64 * 64, 256), 256, 0, stream>>>(Wfg, filter_w, gate_w);
  k_pack_mat<<<CDIV(NL * SCC * RC, 256), 256, 0, stream>>>(Wsk, skip_w, NL * SCC, NL * SCC, RC, RC);
  k_pack_mat<<<CDIV(NL * RC * HC, 256), 256, 0, stream>>>(Wgc, gcn_w, NL * RC, NL * RC, HC, HC);
  k_pack_mat<<<CDIV(ECC * SCC, 256), 256, 0, stream>>>(We1, end1_w, ECC, ECC, SCC, SCC);
  k_pack_mat<<<CDIV(16 * ECC, 256), 256, 0, stream>>>(We2, end2_w, 16, 12, ECC, ECC);
  k_pack_mat<<<CDIV(SCC * PADN, 256), 256, 0, stream>>>(Wli, lin_w, SCC, SCC, PADN, 220);
  k_pack_mat<<<CDIV(NV * PADN, 256), 256, 0, stream>>>(vsb, sa_vs, NV, NV, PADN, NV);

  k_start<<<CDIV(NB * NV * 13 * RC, 256), 256, 0, stream>>>(in, start_w, start_b, cent, xf, xbf);
  hipMemsetAsync(skip, 0, (size_t)POS * SCC * 4, stream);

  // ---- WaveNet layers ----
  int T_in = 13;
  for (int i = 0; i < NL; ++i) {
    int dil = (i % 2 == 0) ? 1 : 2;
    int T_out = T_in - dil;
    int P = POS * T_out;
    // zero hv (pad columns must be zero; T-stride changes every layer)
    hipMemsetAsync(hv, 0, hvRegion * 2 * 2, stream);
    bf16_t* hv0 = hv;
    bf16_t* hv1 = hv + hvRegion;
    k_tcn<<<dim3(P / 16), 32, 0, stream>>>(xbf, T_in, T_out, dil, Wfg + (size_t)i * 64 * 64,
                                           filter_b + i * 32, gate_b + i * 32, h, hv0);
    const bf16_t* Ats[3] = {At0, At1, Atad};
    for (int s = 0; s < 3; ++s) {
      int o1 = 32 + s * 64, o2 = o1 + 32;
      k_hop<<<dim3(2 * T_out, 13, NB), 32, 0, stream>>>(h, hv0, hv1, T_out, o1, Ats[s]);
      k_hop<<<dim3(2 * T_out, 13, NB), 32, 0, stream>>>(h, hv1, nullptr, T_out, o2, Ats[s]);
    }
    // skip conv at last time step only (cropping keeps only t = T_out-1)
    k_gemm<1><<<dim3(POS / 16, SCC / 16), 32, 0, stream>>>(
        h + (size_t)(T_out - 1) * HC, T_out * HC, 0L, Wsk + (size_t)i * SCC * RC, RC, 0L,
        POS, SCC, RC, skip_b + i * SCC, skip, SCC, 0L, nullptr,
        nullptr, nullptr, nullptr, 0, 0, 0, nullptr);
    hipMemsetAsync(stats, 0, 64 * 4, stream);
    k_gemm<4><<<dim3(P / 16, 2), 32, 0, stream>>>(
        h, HC, 0L, Wgc + (size_t)i * RC * HC, HC, 0L,
        P, RC, HC, gcn_b + i * RC, nullptr, 0, 0L, nullptr,
        xf, tmp, stats, T_in, T_out, dil, nullptr);
    int total = P * RC;
    k_bn<<<CDIV(total, 256), 256, 0, stream>>>(tmp, stats, bn_g + i * RC, bn_b + i * RC,
                                               (float)P, total, xf, xbf);
    T_in = T_out;
  }

  // ---- relu(skip) + LayerNorm ----
  k_relu_ln<<<NB, 256, 0, stream>>>(skip, ln_g, ln_b, xln);

  // ---- temporal attention ----
  k_ta1<<<CDIV(1536, 256), 256, 0, stream>>>(in, ta_u1, vec);
  k_ta1b<<<CDIV(NB * TT * NV, 256), 256, 0, stream>>>(vec, ta_u2, lhs_t);
  k_ta2<<<CDIV(NV * NB * TT, 256), 256, 0, stream>>>(in, ta_u3, rhs_t);
  k_ta3<<<NB, 256, 0, stream>>>(lhs_t, rhs_t, ta_be, ta_ve, h_x_sp, tttb);

  // ---- spatial attention ----
  k_sa_lhs<<<CDIV(NB * NV * TT, 256), 256, 0, stream>>>(in, sa_w1, sa_w2, lhs_s);
  k_sa_rhs<<<CDIV(NB * TT * NV, 256), 256, 0, stream>>>(in, sa_w3, rhs_s);
  k_sa2<<<CDIV(NB * NV * PADN, 256), 256, 0, stream>>>(lhs_s, rhs_s, sa_bs, Ptb);
  // Smat[b] = sa_vs @ P[b]  (batched WMMA)
  k_gemm<0><<<dim3(13, 13, NB), 32, 0, stream>>>(
      vsb, PADN, 0L, Ptb, PADN, (long)NV * PADN,
      NV, NV, PADN, nullptr, Smat, NV, (long)NV * NV, nullptr,
      nullptr, nullptr, nullptr, 0, 0, 0, nullptr);
  k_soft_s<<<CDIV(NB * NV, 256), 256, 0, stream>>>(Smat);

  // ---- concat + linear + add ----
  k_concat<<<CDIV(POS * PADN, 256), 256, 0, stream>>>(Smat, tttb, catb);
  k_gemm<0><<<dim3(POS / 16, SCC / 16), 32, 0, stream>>>(
      catb, PADN, 0L, Wli, PADN, 0L, POS, SCC, PADN, lin_b,
      linbuf, SCC, 0L, nullptr, nullptr, nullptr, nullptr, 0, 0, 0, nullptr);
  k_addlin<<<CDIV(POS * SCC, 256), 256, 0, stream>>>(xln, linbuf, t_we, xend);

  // ---- end MLP ----
  k_gemm<2><<<dim3(POS / 16, ECC / 16), 32, 0, stream>>>(
      xend, SCC, 0L, We1, SCC, 0L, POS, ECC, SCC, end1_b,
      nullptr, 0, 0L, e1b, nullptr, nullptr, nullptr, 0, 0, 0, nullptr);
  k_gemm<3><<<dim3(POS / 16, 1), 32, 0, stream>>>(
      e1b, ECC, 0L, We2, ECC, 0L, POS, 12, ECC, end2_b,
      nullptr, 0, 0L, nullptr, nullptr, nullptr, nullptr, 0, 0, 0, dout);
}
